// LlamaTopKAttention_77421080478362
// MI455X (gfx1250) — matrix operand order repaired and checked by reading, not compiled, not run on hardware
//
#include <hip/hip_runtime.h>
#include <hip/hip_bf16.h>
#include <math.h>
#include <stdint.h>

#define NHEADS   32
#define NKVHEADS 8
#define HDIM     128
#define SEQ      2048
#define HID      4096
#define KVDIM    (NKVHEADS * HDIM)   // 1024

typedef __attribute__((ext_vector_type(16))) __bf16 v16bf;
typedef __attribute__((ext_vector_type(8)))  __bf16 v8bf;
typedef __attribute__((ext_vector_type(4)))  __bf16 v4bf;
typedef __attribute__((ext_vector_type(8)))  float  v8f;
typedef __attribute__((ext_vector_type(4)))  int    v4i;

// ---- optional CDNA5 async global->LDS copy (ASYNCcnt-tracked) --------------
#if defined(__has_builtin)
#  if __has_builtin(__builtin_amdgcn_global_load_async_to_lds_b128) && \
      __has_builtin(__builtin_amdgcn_s_wait_asynccnt)
#    define LLMTK_ASYNC_LDS 1
#  endif
#endif
#ifndef LLMTK_ASYNC_LDS
#  define LLMTK_ASYNC_LDS 0
#endif

#if LLMTK_ASYNC_LDS
// Builtin param 0 (per hipcc diagnostic): v4i in address_space(1); LDS side is
// v4i in address_space(3).
typedef __attribute__((address_space(1))) v4i* gbl_v4i_p;
typedef __attribute__((address_space(3))) v4i* lds_v4i_p;
static __device__ __forceinline__ void llmtk_async_cp_b128(const void* gsrc, void* ldst) {
  __builtin_amdgcn_global_load_async_to_lds_b128(
      (gbl_v4i_p)(uintptr_t)gsrc,
      (lds_v4i_p)(uint32_t)(uintptr_t)ldst,
      0, 0);
}
#else
static __device__ __forceinline__ void llmtk_async_cp_b128(const void* gsrc, void* ldst) {
  *(v8bf*)ldst = *(const v8bf*)gsrc;
}
#endif

// ---- fragment loaders ------------------------------------------------------
// A-matrix (16x32 bf16): lanes 0-15 hold K={0..7,16..23}, lanes 16-31 hold
// K={8..15,24..31}; caller pre-offsets p by (lane<16?0:8).
static __device__ __forceinline__ v16bf ld_frag_a(const __bf16* p) {
  v8bf lo = *(const v8bf*)p;
  v8bf hi = *(const v8bf*)(p + 16);
  return __builtin_shufflevector(lo, hi, 0,1,2,3,4,5,6,7,8,9,10,11,12,13,14,15);
}
// B-matrix (32x16 bf16): lane n%16 holds column n; lanes 0-15 K={0..7,16..23},
// lanes 16-31 K={8..15,24..31}; caller pre-offsets p by (lane<16?0:8).
static __device__ __forceinline__ v16bf ld_frag_b(const __bf16* p) {
  v8bf lo = *(const v8bf*)p;
  v8bf hi = *(const v8bf*)(p + 16);
  return __builtin_shufflevector(lo, hi, 0,1,2,3,4,5,6,7,8,9,10,11,12,13,14,15);
}

// ---- fp32 -> bf16 bulk convert --------------------------------------------
__global__ void llmtk_cvt_bf16(const float* __restrict__ in, __bf16* __restrict__ out, int n4) {
  int i = blockIdx.x * blockDim.x + threadIdx.x;
  if (i >= n4) return;
  float4 f = ((const float4*)in)[i];
  v4bf o;
  o[0] = (__bf16)f.x; o[1] = (__bf16)f.y; o[2] = (__bf16)f.z; o[3] = (__bf16)f.w;
  ((v4bf*)out)[i] = o;
}

// ---- bf16 GEMM:  C[M,N] = A[M,K] * B[N,K]^T  (torch Linear layout) ---------
// Block = 256 threads (8 waves).  Block tile 128(M) x 128(N); each wave owns
// 32x64.  K is staged in 64-wide slabs (A+B tiles 16 KB each), double-buffered
// in LDS via async global->LDS copies: 16 WMMAs per commit/barrier.
#define GTK 64
#define TILE_ELEMS (128 * GTK)

static __device__ __forceinline__ void llmtk_stage(
    const __bf16* __restrict__ A, const __bf16* __restrict__ B,
    __bf16* As, __bf16* Bs, int buf,
    int mBlock, int nBlock, int K, int k0, int tid) {
  // 256 threads x 8 chunks of 16B; consecutive lanes copy consecutive chunks.
  const int kc = (tid & 7) * 8;          // 0..56 (elements)
  const int r0 = tid >> 3;               // 0..31
  __bf16* Ad = As + buf * TILE_ELEMS;
  __bf16* Bd = Bs + buf * TILE_ELEMS;
#pragma unroll
  for (int p = 0; p < 4; ++p) {
    const int row = p * 32 + r0;
    llmtk_async_cp_b128(A + (size_t)(mBlock + row) * K + k0 + kc,
                        Ad + row * GTK + kc);
  }
#pragma unroll
  for (int p = 0; p < 4; ++p) {
    const int row = p * 32 + r0;
    llmtk_async_cp_b128(B + (size_t)(nBlock + row) * K + k0 + kc,
                        Bd + row * GTK + kc);
  }
}

static __device__ __forceinline__ void llmtk_commit() {
#if LLMTK_ASYNC_LDS
  __builtin_amdgcn_s_wait_asynccnt(0);
#endif
  __syncthreads();
}

__global__ __launch_bounds__(256)
void llmtk_gemm_bf16(const __bf16* __restrict__ A, const __bf16* __restrict__ B,
                     float* __restrict__ C, int M, int N, int K) {
  __shared__ __bf16 As[2 * TILE_ELEMS];   // 32 KB
  __shared__ __bf16 Bs[2 * TILE_ELEMS];   // 32 KB
  const int tid    = threadIdx.x;
  const int lane   = tid & 31;
  const int wave   = tid >> 5;
  const int waveM  = wave >> 1;           // 0..3 -> 32-row slab
  const int waveN  = wave & 1;            // 0..1 -> 64-col slab
  const int mBlock = blockIdx.y * 128;
  const int nBlock = blockIdx.x * 128;
  const int lrow   = lane & 15;
  const int lhalfA = (lane < 16) ? 0 : 8;
  const int lhalfB = (lane < 16) ? 0 : 8;

  v8f acc[8] = {};

  llmtk_stage(A, B, As, Bs, 0, mBlock, nBlock, K, 0, tid);
  llmtk_commit();

  int cur = 0;
  for (int k0 = 0; k0 < K; k0 += GTK) {
    if (k0 + GTK < K)
      llmtk_stage(A, B, As, Bs, cur ^ 1, mBlock, nBlock, K, k0 + GTK, tid);

    const __bf16* Ab = As + cur * TILE_ELEMS + (waveM * 32 + lrow) * GTK;
    const __bf16* Bb = Bs + cur * TILE_ELEMS + (waveN * 64 + lrow) * GTK;
#pragma unroll
    for (int kk = 0; kk < GTK; kk += 32) {
      // hoist all fragment loads so the scheduler can clause the ds_loads
      v16bf a0 = ld_frag_a(Ab + kk + lhalfA);
      v16bf a1 = ld_frag_a(Ab + 16 * GTK + kk + lhalfA);
      v16bf b0 = ld_frag_b(Bb + kk + lhalfB);
      v16bf b1 = ld_frag_b(Bb + 16 * GTK + kk + lhalfB);
      v16bf b2 = ld_frag_b(Bb + 32 * GTK + kk + lhalfB);
      v16bf b3 = ld_frag_b(Bb + 48 * GTK + kk + lhalfB);
      acc[0] = __builtin_amdgcn_wmma_f32_16x16x32_bf16(false, a0, false, b0, (short)0, acc[0], false, false);
      acc[4] = __builtin_amdgcn_wmma_f32_16x16x32_bf16(false, a1, false, b0, (short)0, acc[4], false, false);
      acc[1] = __builtin_amdgcn_wmma_f32_16x16x32_bf16(false, a0, false, b1, (short)0, acc[1], false, false);
      acc[5] = __builtin_amdgcn_wmma_f32_16x16x32_bf16(false, a1, false, b1, (short)0, acc[5], false, false);
      acc[2] = __builtin_amdgcn_wmma_f32_16x16x32_bf16(false, a0, false, b2, (short)0, acc[2], false, false);
      acc[6] = __builtin_amdgcn_wmma_f32_16x16x32_bf16(false, a1, false, b2, (short)0, acc[6], false, false);
      acc[3] = __builtin_amdgcn_wmma_f32_16x16x32_bf16(false, a0, false, b3, (short)0, acc[3], false, false);
      acc[7] = __builtin_amdgcn_wmma_f32_16x16x32_bf16(false, a1, false, b3, (short)0, acc[7], false, false);
    }
    llmtk_commit();
    cur ^= 1;
  }

  const int nBase = nBlock + waveN * 64 + lrow;
#pragma unroll
  for (int half = 0; half < 2; ++half) {
    const int mTop = mBlock + waveM * 32 + half * 16 + lhalfA;
#pragma unroll
    for (int t = 0; t < 4; ++t)
#pragma unroll
      for (int g = 0; g < 8; ++g)
        C[(size_t)(mTop + g) * N + nBase + t * 16] = acc[half * 4 + t][g];
  }
}

// ---- RoPE + repack [S, nH*128] f32 -> [nH][S][128] bf16 --------------------
__global__ void llmtk_rope_pack(const float* __restrict__ proj, __bf16* __restrict__ out,
                                int nH, int total) {
  int idx = blockIdx.x * blockDim.x + threadIdx.x;      // over S*nH*64
  if (idx >= total) return;
  int i = idx & 63;
  int h = (idx >> 6) % nH;
  int s = idx / (64 * nH);
  float inv = __expf(-(2.0f * (float)i / 128.0f) * __logf(500000.0f));
  float c, sn;
  __sincosf((float)s * inv, &c, &sn);
  const float* row = proj + (size_t)s * (nH * HDIM) + h * HDIM;
  float x1 = row[i], x2 = row[i + 64];
  __bf16* orow = out + ((size_t)h * SEQ + s) * HDIM;
  orow[i]      = (__bf16)(x1 * c - x2 * sn);
  orow[i + 64] = (__bf16)(x2 * c + x1 * sn);
}

// ---- plain repack [S, nH*128] f32 -> [nH][S][128] bf16 (for V) -------------
__global__ void llmtk_pack_bf16(const float* __restrict__ in, __bf16* __restrict__ out,
                                int nH, int total) {
  int idx = blockIdx.x * blockDim.x + threadIdx.x;      // over S*nH*128
  if (idx >= total) return;
  int d = idx & 127;
  int h = (idx >> 7) % nH;
  int s = idx / (128 * nH);
  out[((size_t)h * SEQ + s) * HDIM + d] = (__bf16)in[(size_t)s * (nH * HDIM) + h * HDIM + d];
}

// ---- fused scores + top-k + softmax + V-gather -----------------------------
// One wave per (head, 16-query tile).  Q fragments stay in registers for the
// whole key sweep; 64 keys (16 WMMAs) are scored per barrier pair.
__global__ __launch_bounds__(32)
void llmtk_attn_topk(const __bf16* __restrict__ qb,   // [32][S][128]
                     const __bf16* __restrict__ kb,   // [8][S][128]
                     const __bf16* __restrict__ vb,   // [8][S][128]
                     __bf16* __restrict__ attn,       // [S][4096]
                     const int* __restrict__ topk_ptr) {
  const int h    = blockIdx.x;
  const int qt   = blockIdx.y;
  const int kvh  = h >> 2;                 // groups = 32/8 = 4
  const int lane = threadIdx.x;
  const int lrow   = lane & 15;
  const int lhalfA = (lane < 16) ? 0 : 8;
  const int lhalfB = (lane < 16) ? 0 : 8;

  int KTOP = *topk_ptr;
  if (KTOP > 16) KTOP = 16;
  if (KTOP < 1)  KTOP = 1;

  __shared__ float sc[16 * 64];    // 16 queries x 64 keys
  __shared__ float tv[16][16];     // running top-k values per query row
  __shared__ int   ti[16][16];     // running top-k indices

  float runMin = -3.0e38f;         // per-row threshold (lanes 0-15 only)
  int   runPos = 0;
  if (lane < 16)
    for (int t = 0; t < KTOP; ++t) { tv[lane][t] = -3.0e38f; ti[lane][t] = 0; }

  // Q fragments (A-operand); K-dim 128 = 4 fragments of 32
  const __bf16* qrow = qb + ((size_t)h * SEQ + qt * 16 + lrow) * HDIM + lhalfA;
  v16bf qf0 = ld_frag_a(qrow);
  v16bf qf1 = ld_frag_a(qrow + 32);
  v16bf qf2 = ld_frag_a(qrow + 64);
  v16bf qf3 = ld_frag_a(qrow + 96);

  const float scale = 0.08838834764831845f;  // 1/sqrt(128)

  for (int kt = 0; kt < SEQ / 16; kt += 4) {
    v8f acc[4] = {};
#pragma unroll
    for (int u = 0; u < 4; ++u) {
      const __bf16* krow =
          kb + ((size_t)kvh * SEQ + (kt + u) * 16 + lrow) * HDIM + lhalfB;
      acc[u] = __builtin_amdgcn_wmma_f32_16x16x32_bf16(false, qf0, false, ld_frag_b(krow),      (short)0, acc[u], false, false);
      acc[u] = __builtin_amdgcn_wmma_f32_16x16x32_bf16(false, qf1, false, ld_frag_b(krow + 32), (short)0, acc[u], false, false);
      acc[u] = __builtin_amdgcn_wmma_f32_16x16x32_bf16(false, qf2, false, ld_frag_b(krow + 64), (short)0, acc[u], false, false);
      acc[u] = __builtin_amdgcn_wmma_f32_16x16x32_bf16(false, qf3, false, ld_frag_b(krow + 96), (short)0, acc[u], false, false);
    }

    __syncthreads();   // previous top-k scan done before sc overwrite
#pragma unroll
    for (int u = 0; u < 4; ++u)
#pragma unroll
      for (int g = 0; g < 8; ++g)
        sc[(g + lhalfA) * 64 + u * 16 + lrow] = acc[u][g] * scale;
    __syncthreads();

    if (lane < 16) {   // lane == query row; scan 64 candidates
      for (int j = 0; j < 64; ++j) {
        float s = sc[lane * 64 + j];
        if (s > runMin) {
          tv[lane][runPos] = s;
          ti[lane][runPos] = kt * 16 + j;
          runMin = tv[lane][0]; runPos = 0;
          for (int t = 1; t < KTOP; ++t) {
            float x = tv[lane][t];
            if (x < runMin) { runMin = x; runPos = t; }
          }
        }
      }
    }
  }
  __syncthreads();

  // softmax over the <=16 retained scores (lane per row)
  if (lane < 16) {
    float mx = -3.0e38f;
    for (int t = 0; t < KTOP; ++t) mx = fmaxf(mx, tv[lane][t]);
    float sum = 0.0f;
    for (int t = 0; t < KTOP; ++t) { float e = __expf(tv[lane][t] - mx); tv[lane][t] = e; sum += e; }
    float inv = 1.0f / sum;
    for (int t = 0; t < KTOP; ++t) tv[lane][t] *= inv;
  }
  __syncthreads();

  // weighted gather of top-k V rows; lane owns d = lane*4 .. lane*4+3
  for (int m = 0; m < 16; ++m) {
    float o0 = 0.f, o1 = 0.f, o2 = 0.f, o3 = 0.f;
    for (int t = 0; t < KTOP; ++t) {
      float w = tv[m][t];
      const v4bf vv = *(const v4bf*)(vb + ((size_t)kvh * SEQ + ti[m][t]) * HDIM + lane * 4);
      o0 += w * (float)vv[0];
      o1 += w * (float)vv[1];
      o2 += w * (float)vv[2];
      o3 += w * (float)vv[3];
    }
    v4bf o;
    o[0] = (__bf16)o0; o[1] = (__bf16)o1; o[2] = (__bf16)o2; o[3] = (__bf16)o3;
    *(v4bf*)(attn + (size_t)(qt * 16 + m) * HID + h * HDIM + lane * 4) = o;
  }
}

// ---- host-side orchestration ----------------------------------------------
extern "C" void kernel_launch(void* const* d_in, const int* in_sizes, int n_in,
                              void* d_out, int out_size, void* d_ws, size_t ws_size,
                              hipStream_t stream) {
  (void)in_sizes; (void)n_in; (void)out_size; (void)ws_size;
  const float* hidden = (const float*)d_in[0];
  const float* Wq     = (const float*)d_in[1];
  const float* Wk     = (const float*)d_in[2];
  const float* Wv     = (const float*)d_in[3];
  const float* Wo     = (const float*)d_in[4];
  const int*   topk   = (const int*)d_in[5];
  float* out = (float*)d_out;

  // Workspace layout (~151 MB with aliasing; lifetimes are disjoint).
  char* ws = (char*)d_ws;
  size_t off = 0;
  auto walloc = [&](size_t bytes) -> char* {
    char* p = ws + off;
    off += (bytes + 255) & ~(size_t)255;
    return p;
  };
  __bf16* hbf = (__bf16*)walloc((size_t)SEQ * HID * 2);
  __bf16* wqb = (__bf16*)walloc((size_t)HID * HID * 2);
  __bf16* wkb = (__bf16*)walloc((size_t)KVDIM * HID * 2);
  __bf16* wvb = (__bf16*)walloc((size_t)KVDIM * HID * 2);
  __bf16* wob = (__bf16*)walloc((size_t)HID * HID * 2);
  float*  qf  = (float*)walloc((size_t)SEQ * HID * 4);
  float*  kf  = (float*)walloc((size_t)SEQ * KVDIM * 4);
  float*  vf  = (float*)walloc((size_t)SEQ * KVDIM * 4);
  // Aliases (region dead by the time the alias is written):
  __bf16* qbf = wqb;            // RoPE'd Q  [32][S][128]  (Wq bf16 dead after Q GEMM)
  __bf16* kbf = wkb;            // RoPE'd K  [8][S][128]
  __bf16* vbf = wvb;            // packed V  [8][S][128]
  __bf16* abf = (__bf16*)qf;    // attention out bf16 [S][4096] (q_f32 dead after RoPE)

  auto cvt = [&](const float* src, __bf16* dst, size_t n) {
    int n4 = (int)(n / 4);
    llmtk_cvt_bf16<<<(n4 + 255) / 256, 256, 0, stream>>>(src, dst, n4);
  };
  cvt(hidden, hbf, (size_t)SEQ * HID);
  cvt(Wq,     wqb, (size_t)HID * HID);
  cvt(Wk,     wkb, (size_t)KVDIM * HID);
  cvt(Wv,     wvb, (size_t)KVDIM * HID);
  cvt(Wo,     wob, (size_t)HID * HID);

  // Projections: x @ W.T
  llmtk_gemm_bf16<<<dim3(HID / 128,   SEQ / 128), 256, 0, stream>>>(hbf, wqb, qf, SEQ, HID,   HID);
  llmtk_gemm_bf16<<<dim3(KVDIM / 128, SEQ / 128), 256, 0, stream>>>(hbf, wkb, kf, SEQ, KVDIM, HID);
  llmtk_gemm_bf16<<<dim3(KVDIM / 128, SEQ / 128), 256, 0, stream>>>(hbf, wvb, vf, SEQ, KVDIM, HID);

  // RoPE + head-major repack (q overwrites wqb, k overwrites wkb -> safe)
  {
    int tq = SEQ * NHEADS * 64;
    llmtk_rope_pack<<<(tq + 255) / 256, 256, 0, stream>>>(qf, qbf, NHEADS, tq);
    int tk = SEQ * NKVHEADS * 64;
    llmtk_rope_pack<<<(tk + 255) / 256, 256, 0, stream>>>(kf, kbf, NKVHEADS, tk);
    int tvn = SEQ * NKVHEADS * HDIM;
    llmtk_pack_bf16<<<(tvn + 255) / 256, 256, 0, stream>>>(vf, vbf, NKVHEADS, tvn);
  }

  // Fused top-k attention -> bf16 attention matrix (overwrites q_f32 region)
  llmtk_attn_topk<<<dim3(NHEADS, SEQ / 16), 32, 0, stream>>>(qbf, kbf, vbf, abf, topk);

  // Output projection: attn @ Wo.T -> f32 result
  llmtk_gemm_bf16<<<dim3(HID / 128, SEQ / 128), 256, 0, stream>>>(abf, wob, out, SEQ, HID, HID);
}
